// TTCN_3496103379118
// MI455X (gfx1250) — matrix operand order repaired and checked by reading, not compiled, not run on hardware
//
#include <hip/hip_runtime.h>

// ---------------------------------------------------------------------------
// TTCN fused kernels for MI455X (gfx1250, wave32, WMMA bf16 16x16x32)
//   D_IN = 128, D_H = 128, N = 64, L = 128
// Kernel 1: H = relu(relu(X@W1+b1)@W2+b2)  -> bf16 workspace (2 MB)
// Kernel 2: per (n,h): filt = H@W3[:,h*128:+128]+b3 ; masked softmax over L ;
//           out[n,h] = relu( sum_{l,f} x[n,l,f]*p[l,f] + t_bias[h] )
// ---------------------------------------------------------------------------

typedef __attribute__((ext_vector_type(16))) __bf16        v16bf;
typedef __attribute__((ext_vector_type(8)))  float         v8f;
typedef __attribute__((ext_vector_type(8)))  unsigned int  v8u;

#define D_IN   128
#define D_HID  128
#define NBAT   64
#define LSEQ   128
#define S16    136   // bf16 tile stride (128 + 8 pad) -> conflict-free frags
#define SF     132   // f32 filt tile stride (128 + 4 pad)
#define NEGV   (-100000000.0f)
#define HPB    8     // h-values per workgroup in kernel 2

#define SMEM1  (3 * 128 * S16 * 2)                               // 104448 B
#define SMEM2  (2 * 128 * S16 * 2 + 128 * SF * 4 + 512)          // 137728 B

// round-to-nearest-even f32 -> bf16 (raw u16)
__device__ __forceinline__ unsigned short f2bf(float x) {
  unsigned int u = __builtin_bit_cast(unsigned int, x);
  u += 0x7FFFu + ((u >> 16) & 1u);
  return (unsigned short)(u >> 16);
}

// A fragment (16x32 bf16, row-major LDS tile, stride S16):
// lane = m + 16*g : VGPR0..7 hold K = 16g+0..15 (contiguous)
__device__ __forceinline__ v16bf load_a_frag(const unsigned short* lds,
                                             int rowBase, int kBase, int lane) {
  int m = lane & 15, g = lane >> 4;
  const unsigned short* p = lds + (rowBase + m) * S16 + kBase + 16 * g;
  uint4 c0 = *(const uint4*)(p);
  uint4 c1 = *(const uint4*)(p + 8);
  v8u r = {c0.x, c0.y, c0.z, c0.w, c1.x, c1.y, c1.z, c1.w};
  return __builtin_bit_cast(v16bf, r);
}

// B fragment (32x16 bf16) from TRANSPOSED LDS tile Wt[col][k], stride S16:
// lane = n + 16*g : VGPR v holds K = 16g+2v, 16g+2v+1 -> 32 contiguous bytes
__device__ __forceinline__ v16bf load_b_frag(const unsigned short* ldsT,
                                             int colBase, int kBase, int lane) {
  int nn = lane & 15, g = lane >> 4;
  const unsigned short* p = ldsT + (colBase + nn) * S16 + kBase + 16 * g;
  uint4 c0 = *(const uint4*)(p);
  uint4 c1 = *(const uint4*)(p + 8);
  v8u r = {c0.x, c0.y, c0.z, c0.w, c1.x, c1.y, c1.z, c1.w};
  return __builtin_bit_cast(v16bf, r);
}

// ---------------------------------------------------------------------------
// Kernel 1: two fused 128x128x128 bf16 WMMA GEMMs (filter-generator MLP body)
// one workgroup per batch element n; 256 threads = 8 waves, 16 rows per wave
// ---------------------------------------------------------------------------
__global__ __launch_bounds__(256)
void ttcn_mlp_kernel(const float* __restrict__ x,
                     const float* __restrict__ W1, const float* __restrict__ b1,
                     const float* __restrict__ W2, const float* __restrict__ b2,
                     unsigned short* __restrict__ Hws) {
  extern __shared__ unsigned char smem[];
  unsigned short* Ab = (unsigned short*)smem;        // X      bf16 [128][S16]
  unsigned short* Bt = Ab + 128 * S16;               // W^T    bf16 [128][S16]
  unsigned short* Cb = Bt + 128 * S16;               // hidden bf16 [128][S16]

  const int n = blockIdx.x, tid = threadIdx.x;
  const int lane = tid & 31, wave = tid >> 5;
  const int rowBase = wave * 16;

  for (int idx = tid; idx < 128 * 128; idx += 256) {
    int l = idx >> 7, k = idx & 127;
    Ab[l * S16 + k] = f2bf(x[(n * LSEQ + l) * D_IN + k]);
  }
  for (int idx = tid; idx < 128 * 128; idx += 256) {
    int k = idx >> 7, j = idx & 127;
    Bt[j * S16 + k] = f2bf(W1[k * D_HID + j]);       // store W1 transposed
  }
  __syncthreads();

  { // layer 1: Cb = relu(X @ W1 + b1)
    v16bf a[4];
    for (int kk = 0; kk < 4; ++kk) a[kk] = load_a_frag(Ab, rowBase, kk * 32, lane);
    for (int ct = 0; ct < 8; ++ct) {
      v8f acc = {};
      for (int kk = 0; kk < 4; ++kk) {
        v16bf b = load_b_frag(Bt, ct * 16, kk * 32, lane);
        acc = __builtin_amdgcn_wmma_f32_16x16x32_bf16(false, a[kk], false, b,
                                                      (short)0, acc, false, false);
      }
      int col = ct * 16 + (lane & 15), g = lane >> 4;
      float bv = b1[col];
      for (int v = 0; v < 8; ++v)
        Cb[(rowBase + v + 8 * g) * S16 + col] = f2bf(fmaxf(acc[v] + bv, 0.0f));
    }
  }
  __syncthreads();

  for (int idx = tid; idx < 128 * 128; idx += 256) {
    int k = idx >> 7, j = idx & 127;
    Bt[j * S16 + k] = f2bf(W2[k * D_HID + j]);       // store W2 transposed
  }
  __syncthreads();

  { // layer 2: H = relu(Cb @ W2 + b2) -> global bf16 workspace
    v16bf a[4];
    for (int kk = 0; kk < 4; ++kk) a[kk] = load_a_frag(Cb, rowBase, kk * 32, lane);
    for (int ct = 0; ct < 8; ++ct) {
      v8f acc = {};
      for (int kk = 0; kk < 4; ++kk) {
        v16bf b = load_b_frag(Bt, ct * 16, kk * 32, lane);
        acc = __builtin_amdgcn_wmma_f32_16x16x32_bf16(false, a[kk], false, b,
                                                      (short)0, acc, false, false);
      }
      int col = ct * 16 + (lane & 15), g = lane >> 4;
      float bv = b2[col];
      for (int v = 0; v < 8; ++v)
        Hws[(size_t)(n * LSEQ + rowBase + v + 8 * g) * D_HID + col] =
            f2bf(fmaxf(acc[v] + bv, 0.0f));
    }
  }
}

// ---------------------------------------------------------------------------
// Kernel 2: dominant GEMM + masked softmax over L + contraction with x
// grid (64, 16); each WG keeps H_n in LDS and walks HPB=8 h-blocks of W3
// ---------------------------------------------------------------------------
__global__ __launch_bounds__(256)
void ttcn_filter_kernel(const float* __restrict__ x,
                        const float* __restrict__ mask,
                        const float* __restrict__ W3,
                        const float* __restrict__ b3,
                        const float* __restrict__ t_bias,
                        const unsigned short* __restrict__ Hws,
                        float* __restrict__ out) {
  extern __shared__ unsigned char smem[];
  unsigned short* Hb = (unsigned short*)smem;                 // bf16 [128][S16]
  unsigned short* Wt = Hb + 128 * S16;                        // bf16 [128][S16]
  float* filt = (float*)(smem + (size_t)2 * 128 * S16 * 2);   // f32  [128][SF]
  float* red  = filt + 128 * SF;                              // f32  [128]

  const int n = blockIdx.x, hb = blockIdx.y;
  const int tid = threadIdx.x, lane = tid & 31, wave = tid >> 5;
  const int rowBase = wave * 16;

  for (int idx = tid; idx < 128 * 128; idx += 256) {
    int l = idx >> 7, k = idx & 127;
    Hb[l * S16 + k] = Hws[(size_t)(n * LSEQ + l) * D_HID + k];
  }

  for (int it = 0; it < HPB; ++it) {
    const int hh = hb * HPB + it;
    __syncthreads();  // prior iteration done with Wt/filt/red (and Hb loaded)

    // stage W3[:, hh*128 : +128] transposed as bf16; prefetch next block
    for (int idx = tid; idx < 128 * 128; idx += 256) {
      int k = idx >> 7, f = idx & 127;
      Wt[f * S16 + k] = f2bf(W3[(size_t)k * 16384 + hh * 128 + f]);
    }
    if (it + 1 < HPB && tid < 128)
      __builtin_prefetch(&W3[(size_t)tid * 16384 + (hh + 1) * 128], 0, 0);
    __syncthreads();

    // filt = H @ W3_blk + b3   (128x128x128, 32 WMMA per wave)
    {
      v16bf a[4];
      for (int kk = 0; kk < 4; ++kk) a[kk] = load_a_frag(Hb, rowBase, kk * 32, lane);
      for (int ct = 0; ct < 8; ++ct) {
        v8f acc = {};
        for (int kk = 0; kk < 4; ++kk) {
          v16bf b = load_b_frag(Wt, ct * 16, kk * 32, lane);
          acc = __builtin_amdgcn_wmma_f32_16x16x32_bf16(false, a[kk], false, b,
                                                        (short)0, acc, false, false);
        }
        int col = ct * 16 + (lane & 15), g = lane >> 4;
        float bv = b3[hh * 128 + col];
        for (int v = 0; v < 8; ++v)
          filt[(rowBase + v + 8 * g) * SF + col] = acc[v] + bv;
      }
    }
    __syncthreads();

    // masked softmax over L per column f, fused with sum_l x[l,f]*p[l,f]
    if (tid < 128) {
      const int f = tid;
      float mx = -3.4e38f;
      for (int l = 0; l < 128; ++l) {
        float m = mask[n * LSEQ + l];
        float v = filt[l * SF + f] * m + (1.0f - m) * NEGV;
        mx = fmaxf(mx, v);
      }
      float num = 0.0f, den = 0.0f;
      for (int l = 0; l < 128; ++l) {
        float m = mask[n * LSEQ + l];
        float v = filt[l * SF + f] * m + (1.0f - m) * NEGV;
        float e = __expf(v - mx);
        den += e;
        num += e * x[(n * LSEQ + l) * D_IN + f];
      }
      red[f] = num / den;
    }
    __syncthreads();
    for (int off = 64; off > 0; off >>= 1) {
      if (tid < off) red[tid] += red[tid + off];
      __syncthreads();
    }
    if (tid == 0)
      out[n * D_HID + hh] = fmaxf(red[0] + t_bias[hh], 0.0f);
  }
}

// ---------------------------------------------------------------------------
extern "C" void kernel_launch(void* const* d_in, const int* in_sizes, int n_in,
                              void* d_out, int out_size, void* d_ws, size_t ws_size,
                              hipStream_t stream) {
  (void)in_sizes; (void)n_in; (void)out_size; (void)ws_size;
  const float* x    = (const float*)d_in[0];
  const float* mask = (const float*)d_in[1];
  const float* W1   = (const float*)d_in[2];
  const float* b1   = (const float*)d_in[3];
  const float* W2   = (const float*)d_in[4];
  const float* b2   = (const float*)d_in[5];
  const float* W3   = (const float*)d_in[6];
  const float* b3   = (const float*)d_in[7];
  const float* tb   = (const float*)d_in[8];
  unsigned short* Hws = (unsigned short*)d_ws;   // 64*128*128 bf16 = 2 MB
  float* out = (float*)d_out;                    // (64, 128) f32

  hipFuncSetAttribute((const void*)ttcn_mlp_kernel,
                      hipFuncAttributeMaxDynamicSharedMemorySize, SMEM1);
  hipFuncSetAttribute((const void*)ttcn_filter_kernel,
                      hipFuncAttributeMaxDynamicSharedMemorySize, SMEM2);

  ttcn_mlp_kernel<<<dim3(NBAT), dim3(256), SMEM1, stream>>>(x, W1, b1, W2, b2, Hws);
  ttcn_filter_kernel<<<dim3(NBAT, D_HID / HPB), dim3(256), SMEM2, stream>>>(
      x, mask, W3, b3, tb, Hws, out);
}
